// Illum_Guided_TransformerBlock_9165460209674
// MI455X (gfx1250) — matrix operand order repaired and checked by reading, not compiled, not run on hardware
//
#include <hip/hip_runtime.h>
#include <math.h>

typedef __attribute__((ext_vector_type(16))) __bf16 v16bf;
typedef __attribute__((ext_vector_type(8)))  float  v8f;
typedef __attribute__((ext_vector_type(4))) unsigned int u32x4;
typedef __attribute__((ext_vector_type(8))) int          i32x8;
typedef __attribute__((ext_vector_type(4))) int          i32x4;

#define HH 128
#define WW 128
#define HW 16384

__device__ __forceinline__ float bf2f(unsigned short u) {
    return __uint_as_float(((unsigned)u) << 16);
}
__device__ __forceinline__ unsigned short f2bf(float f) {
    unsigned u = __float_as_uint(f);
    u += 0x7FFFu + ((u >> 16) & 1u);            // round-to-nearest-even
    return (unsigned short)(u >> 16);
}

// 16-bit A-matrix 16x32 wave32 fragment: element i of the v16bf
// fragment on this lane corresponds to local K = 16*(lane/16) + i (matches B):
__device__ __forceinline__ int a_klocal(int lane, int i) {
    int base = (lane & 16) ? 16 : 0;
    return base + i;
}
// B-matrix 32x16: lane%16 = column N, lanes 0-15 hold K=0..15, lanes 16-31 K=16..31.

// ---------------------------------------------------------------------------
// Implicit-GEMM conv:  out[b][m][p] = sum_k A[m][k] * im2col(B)[k][p]
// Block tile: 64 (M) x 128 (N = one image row).  8 waves, each wave owns 16
// columns and 4 M-subtiles -> 4 WMMAs per K-chunk reusing one B fragment.
// A panel (64x32 from row-major [M,K], stride K) is staged by the Tensor Data
// Mover when the toolchain exposes it (TENSORcnt-tracked 2D tile DMA, with
// hardware OOB zero-fill standing in for the guard rows); otherwise manually.
// taps==1 : 1x1 conv, k == input channel.
// taps==9 : full 3x3 conv with TAP-MAJOR k = t*inC + c (weights pre-reordered),
//           so each B row is one contiguous dx-shifted image row, zero padded.
// act==1  : exact GELU.  res (f32, optional): residual added post-activation.
// ---------------------------------------------------------------------------
__global__ __launch_bounds__(256)
void gemm_conv_wmma(const unsigned short* __restrict__ A,
                    const unsigned short* __restrict__ B,
                    const float* __restrict__ res,
                    float* __restrict__ outF,
                    unsigned short* __restrict__ outB,
                    int M, int K, int inC, int taps, int act)
{
    __shared__ __align__(16) unsigned short sA[64 * 32];   // 4 KB
    __shared__ __align__(16) unsigned short sB[32 * 128];  // 8 KB
    const int lane  = threadIdx.x & 31;
    const int wid   = threadIdx.x >> 5;
    const int mt    = blockIdx.y;                  // 64-row M tile
    const int ntile = blockIdx.x;                  // image row (N tile of 128)
    const int bz    = blockIdx.z;
    const int pbase = ntile * 128;
    const int arow  = lane & 15;
    const int ncol  = wid * 16 + (lane & 15);      // this wave's output column
    const int kb    = (lane & 16) ? 16 : 0;

    v8f acc0 = {}, acc1 = {}, acc2 = {}, acc3 = {};

    for (int kk = 0; kk < K; kk += 32) {
        // ---- stage A panel: 64 rows x 32 k ----
#if __has_builtin(__builtin_amdgcn_tensor_load_to_lds)
        if (wid == 0) {
            // Tensor DMA descriptor (CDNA5 ISA):
            // 2D tile: tile_dim0=32 (k), tile_dim1=64 (rows), dim0 stride = K.
            // tensor_dim0/1 = remaining k/rows so OOB reads return zero.
            unsigned long long ga =
                (unsigned long long)(const void*)(A + (size_t)mt * 64 * K + kk);
            unsigned ldsa = (unsigned)(unsigned long long)(void*)sA;
            int remK = K - kk;
            int remM = M - mt * 64;
            u32x4 g0;
            g0[0] = 1u;                                  // count=1 (valid, user)
            g0[1] = ldsa;                                // lds_addr
            g0[2] = (unsigned)ga;                        // global_addr[31:0]
            g0[3] = (unsigned)((ga >> 32) & 0x01FFFFFFu) // global_addr[56:32]
                  | 0x80000000u;                         // type=2 ("image")
            i32x8 g1;
            g1[0] = 0x00010000;                          // data_size = 2 bytes
            g1[1] = (int)((unsigned)(remK & 0xFFFF) << 16);        // dim0 lo16
            g1[2] = (int)(((unsigned)(remK >> 16) & 0xFFFFu)       // dim0 hi16
                  | ((unsigned)(remM & 0xFFFF) << 16));            // dim1 lo16
            g1[3] = (int)(((unsigned)(remM >> 16) & 0xFFFFu)       // dim1 hi16
                  | (32u << 16));                                  // tile_dim0
            g1[4] = 64;                                  // tile_dim1 (tile_dim2=0)
            g1[5] = K;                                   // dim0_stride lo32
            g1[6] = 0;
            g1[7] = 0;
            i32x4 gz = {0, 0, 0, 0};                     // groups 2/3: 2D tensor
#if defined(__clang_major__) && (__clang_major__ >= 23)
            i32x8 g4 = {0, 0, 0, 0, 0, 0, 0, 0};
            __builtin_amdgcn_tensor_load_to_lds(g0, g1, gz, gz, g4, 0);
#else
            __builtin_amdgcn_tensor_load_to_lds(g0, g1, gz, gz, 0);
#endif
#if __has_builtin(__builtin_amdgcn_s_wait_tensorcnt)
            __builtin_amdgcn_s_wait_tensorcnt(0);
#else
            asm volatile("s_wait_tensorcnt 0x0" ::: "memory");
#endif
        }
#else
        for (int e = threadIdx.x; e < 2048; e += 256) {
            int rl = e >> 5, kl = e & 31;
            int gr = mt * 64 + rl, gk = kk + kl;
            sA[e] = (gr < M && gk < K) ? A[(size_t)gr * K + gk] : (unsigned short)0;
        }
        __builtin_prefetch(A + (size_t)(mt * 64) * K + kk + 32, 0, 1);
#endif
        // ---- stage B tile: 32 k-rows x 128 cols ----
        if (taps == 1) {
            // each k-row is a contiguous 256B image row: b128 loads/stores
            for (int e = threadIdx.x; e < 512; e += 256) {
                int r = e >> 4, c8 = (e & 15) * 8;
                int k = kk + r;
                uint4 vv = {0u, 0u, 0u, 0u};
                if (k < K)
                    vv = *(const uint4*)(B + ((size_t)bz * inC + k) * HW + pbase + c8);
                *(uint4*)(&sB[r * 128 + c8]) = vv;
            }
        } else {
            for (int e = threadIdx.x; e < 4096; e += 256) {
                int r = e >> 7, col = e & 127;
                int k = kk + r;
                unsigned short val = 0;
                if (k < K) {
                    int t = k / inC, c = k - t * inC;       // tap-major
                    int yy = ntile + t / 3 - 1;
                    int xx = col + (t % 3) - 1;
                    if (yy >= 0 && yy < HH && xx >= 0 && xx < WW)
                        val = B[((size_t)bz * inC + c) * HW + yy * WW + xx];
                }
                sB[r * 128 + col] = val;
            }
        }
        __syncthreads();

        union { v16bf v; unsigned short u[16]; } bfr, af;
        #pragma unroll
        for (int i = 0; i < 16; ++i)
            bfr.u[i] = sB[(kb + i) * 128 + ncol];

        #pragma unroll
        for (int i = 0; i < 16; ++i) af.u[i] = sA[(0 * 16 + arow) * 32 + a_klocal(lane, i)];
        acc0 = __builtin_amdgcn_wmma_f32_16x16x32_bf16(false, af.v, false, bfr.v, (short)0, acc0, false, false);
        #pragma unroll
        for (int i = 0; i < 16; ++i) af.u[i] = sA[(1 * 16 + arow) * 32 + a_klocal(lane, i)];
        acc1 = __builtin_amdgcn_wmma_f32_16x16x32_bf16(false, af.v, false, bfr.v, (short)0, acc1, false, false);
        #pragma unroll
        for (int i = 0; i < 16; ++i) af.u[i] = sA[(2 * 16 + arow) * 32 + a_klocal(lane, i)];
        acc2 = __builtin_amdgcn_wmma_f32_16x16x32_bf16(false, af.v, false, bfr.v, (short)0, acc2, false, false);
        #pragma unroll
        for (int i = 0; i < 16; ++i) af.u[i] = sA[(3 * 16 + arow) * 32 + a_klocal(lane, i)];
        acc3 = __builtin_amdgcn_wmma_f32_16x16x32_bf16(false, af.v, false, bfr.v, (short)0, acc3, false, false);
        __syncthreads();
    }

    const int mb = (lane & 16) ? 8 : 0;            // C/D layout: VGPR r -> M = mb + r
    const int p  = pbase + ncol;
    #pragma unroll
    for (int ms = 0; ms < 4; ++ms) {
        v8f acc = (ms == 0) ? acc0 : (ms == 1) ? acc1 : (ms == 2) ? acc2 : acc3;
        #pragma unroll
        for (int r = 0; r < 8; ++r) {
            int mrow = mt * 64 + ms * 16 + mb + r;
            if (mrow < M) {
                float v = acc[r];
                if (act) v = 0.5f * v * (1.0f + erff(v * 0.70710678118f));
                size_t oi = ((size_t)bz * M + mrow) * HW + p;
                if (res)  v += res[oi];
                if (outF) outF[oi] = v;
                if (outB) outB[oi] = f2bf(v);
            }
        }
    }
}

// ---- LayerNorm over channels per (b, spatial) position, f32 -> bf16 -------
__global__ __launch_bounds__(256)
void ln_bf16_kernel(const float* __restrict__ x, const float* __restrict__ gw,
                    const float* __restrict__ gb, unsigned short* __restrict__ out, int C)
{
    int gid = blockIdx.x * blockDim.x + threadIdx.x;      // over 2*HW positions
    int b = gid / HW, p = gid - b * HW;
    const float* xp = x + (size_t)b * C * HW + p;
    float s = 0.f, s2 = 0.f;
    for (int c = 0; c < C; ++c) { float v = xp[(size_t)c * HW]; s += v; s2 += v * v; }
    float mu = s / C;
    float var = fmaxf(s2 / C - mu * mu, 0.0f);
    float rs = rsqrtf(var + 1e-5f);
    unsigned short* op = out + (size_t)b * C * HW + p;
    for (int c = 0; c < C; ++c)
        op[(size_t)c * HW] = f2bf((xp[(size_t)c * HW] - mu) * rs * gw[c] + gb[c]);
}

// ---- depthwise 3x3 (groups = C), bf16 in/out, f32 weights -----------------
__global__ __launch_bounds__(256)
void dwconv3x3_kernel(const unsigned short* __restrict__ in,
                      const float* __restrict__ w,
                      unsigned short* __restrict__ out, int C)
{
    size_t gid = (size_t)blockIdx.x * blockDim.x + threadIdx.x;
    int p = (int)(gid % HW);
    size_t bc = gid / HW;
    int c = (int)(bc % C);
    int py = p >> 7, px = p & 127;
    const unsigned short* ip = in + bc * HW;
    float s = 0.f;
    #pragma unroll
    for (int t = 0; t < 9; ++t) {
        int y = py + t / 3 - 1, x = px + t % 3 - 1;
        if (y >= 0 && y < HH && x >= 0 && x < WW)
            s += bf2f(ip[y * WW + x]) * w[c * 9 + t];
    }
    out[gid] = f2bf(s);
}

// ---- L2-normalize one bf16 row of length HW (in place) --------------------
__global__ __launch_bounds__(256)
void rownorm_kernel(unsigned short* __restrict__ q, int mod, int stride)
{
    __shared__ float red[256];
    int r = (blockIdx.x / mod) * stride + (blockIdx.x % mod);
    unsigned short* row = q + (size_t)r * HW;
    float s = 0.f;
    for (int i = threadIdx.x; i < HW; i += 256) { float v = bf2f(row[i]); s += v * v; }
    red[threadIdx.x] = s; __syncthreads();
    for (int o = 128; o > 0; o >>= 1) {
        if (threadIdx.x < o) red[threadIdx.x] += red[threadIdx.x + o];
        __syncthreads();
    }
    float scale = 1.0f / fmaxf(sqrtf(red[0]), 1e-12f);
    for (int i = threadIdx.x; i < HW; i += 256)
        row[i] = f2bf(bf2f(row[i]) * scale);
}

// ---- attn = q_n . k_n^T  (48x48 per batch-head, K = HW) -------------------
__global__ __launch_bounds__(32)
void qk_wmma_kernel(const unsigned short* __restrict__ qn,
                    const unsigned short* __restrict__ kvn,
                    float* __restrict__ attn)
{
    int mt = blockIdx.y, ntile = blockIdx.x, bh = blockIdx.z;
    int lane = threadIdx.x;
    int qoff = (bh >> 2) * 192 + (bh & 3) * 48;           // q rows in (b,192,HW)
    int koff = (bh >> 2) * 384 + (bh & 3) * 48;           // k rows in (b,384,HW)
    int arow = mt * 16 + (lane & 15);
    int bcol = ntile * 16 + (lane & 15);
    int kb = (lane & 16) ? 16 : 0;
    v8f acc = {};
    for (int kk = 0; kk < HW; kk += 32) {
        union { v16bf v; unsigned short u[16]; } af, bfr;
        #pragma unroll
        for (int i = 0; i < 16; ++i)
            af.u[i] = qn[(size_t)(qoff + arow) * HW + kk + a_klocal(lane, i)];
        #pragma unroll
        for (int i = 0; i < 16; ++i)
            bfr.u[i] = kvn[(size_t)(koff + bcol) * HW + kk + kb + i];
        acc = __builtin_amdgcn_wmma_f32_16x16x32_bf16(false, af.v, false, bfr.v,
                                                      (short)0, acc, false, false);
    }
    int mb = (lane & 16) ? 8 : 0;
    #pragma unroll
    for (int r = 0; r < 8; ++r) {
        int m = mt * 16 + mb + r;
        int n = ntile * 16 + (lane & 15);
        attn[(size_t)bh * 2304 + m * 48 + n] = acc[r];
    }
}

// ---- softmax(attn * temp) rows of 48, f32 -> bf16 -------------------------
__global__ void softmax_temp_kernel(const float* __restrict__ attn,
                                    const float* __restrict__ temp,
                                    unsigned short* __restrict__ out)
{
    int tid = blockIdx.x * blockDim.x + threadIdx.x;
    if (tid >= 384) return;
    int bh = tid / 48, r = tid % 48;
    float t = temp[bh & 3];
    const float* row = attn + (size_t)bh * 2304 + r * 48;
    float mx = -1e30f;
    for (int j = 0; j < 48; ++j) mx = fmaxf(mx, row[j] * t);
    float s = 0.f;
    for (int j = 0; j < 48; ++j) s += __expf(row[j] * t - mx);
    float inv = 1.0f / s;
    unsigned short* o = out + (size_t)bh * 2304 + r * 48;
    for (int j = 0; j < 48; ++j) o[j] = f2bf(__expf(row[j] * t - mx) * inv);
}

// ---- out = attn . v  (48 x HW per batch-head, K = 48 padded to 64) --------
__global__ __launch_bounds__(32)
void av_wmma_kernel(const unsigned short* __restrict__ attn,
                    const unsigned short* __restrict__ kv,
                    unsigned short* __restrict__ out)
{
    int nt = blockIdx.x, mt = blockIdx.y, bh = blockIdx.z;
    int lane = threadIdx.x;
    int voff = (bh >> 2) * 384 + 192 + (bh & 3) * 48;     // v rows in (b,384,HW)
    int ooff = (bh >> 2) * 192 + (bh & 3) * 48;           // out rows in (b,192,HW)
    int arow = mt * 16 + (lane & 15);
    int p = nt * 16 + (lane & 15);
    int kb = (lane & 16) ? 16 : 0;
    v8f acc = {};
    for (int kk = 0; kk < 64; kk += 32) {
        union { v16bf v; unsigned short u[16]; } af, bfr;
        #pragma unroll
        for (int i = 0; i < 16; ++i) {
            int k = kk + a_klocal(lane, i);
            af.u[i] = (k < 48) ? attn[(size_t)bh * 2304 + arow * 48 + k] : (unsigned short)0;
        }
        #pragma unroll
        for (int i = 0; i < 16; ++i) {
            int k = kk + kb + i;
            bfr.u[i] = (k < 48) ? kv[(size_t)(voff + k) * HW + p] : (unsigned short)0;
        }
        acc = __builtin_amdgcn_wmma_f32_16x16x32_bf16(false, af.v, false, bfr.v,
                                                      (short)0, acc, false, false);
    }
    int mb = (lane & 16) ? 8 : 0;
    #pragma unroll
    for (int r = 0; r < 8; ++r) {
        int m = mt * 16 + mb + r;
        out[(size_t)(ooff + m) * HW + p] = f2bf(acc[r]);
    }
}

__global__ void cvt_f32_bf16_kernel(const float* __restrict__ s,
                                    unsigned short* __restrict__ d, int n)
{
    int i = blockIdx.x * blockDim.x + threadIdx.x;
    if (i < n) d[i] = f2bf(s[i]);
}

// OIHW f32 3x3 weights -> bf16 TAP-MAJOR: dst[m][t*inC + c] = src[m][c][t]
__global__ void cvt3x3_tapmajor_kernel(const float* __restrict__ s,
                                       unsigned short* __restrict__ d,
                                       int inC, int n)
{
    int i = blockIdx.x * blockDim.x + threadIdx.x;
    if (i >= n) return;
    int m = i / (9 * inC);
    int r = i - m * (9 * inC);
    int t = r / inC;
    int c = r - t * inC;
    d[i] = f2bf(s[((size_t)m * inC + c) * 9 + t]);
}

// ---------------------------------------------------------------------------
extern "C" void kernel_launch(void* const* d_in, const int* in_sizes, int n_in,
                              void* d_out, int out_size, void* d_ws, size_t ws_size,
                              hipStream_t stream)
{
    const float* x      = (const float*)d_in[0];
    const float* illum  = (const float*)d_in[1];
    const float* ln1_w  = (const float*)d_in[2];
    const float* ln1_b  = (const float*)d_in[3];
    const float* lnL_w  = (const float*)d_in[4];
    const float* lnL_b  = (const float*)d_in[5];
    const float* ln2_w  = (const float*)d_in[6];
    const float* ln2_b  = (const float*)d_in[7];
    const float* q_w    = (const float*)d_in[8];
    const float* qdw_w  = (const float*)d_in[9];
    const float* kv_w   = (const float*)d_in[10];
    const float* kvdw_w = (const float*)d_in[11];
    const float* temp   = (const float*)d_in[12];
    const float* proj_w = (const float*)d_in[13];
    const float* pin_w  = (const float*)d_in[14];
    const float* dw_w   = (const float*)d_in[15];
    const float* pout_w = (const float*)d_in[16];

    const int B = 2, C = 192, HID = 510;
    const size_t NCHW = (size_t)B * C * HW;               // 6,291,456
    const size_t NKV  = (size_t)B * 2 * C * HW;
    const size_t NHID = (size_t)B * HID * HW;

    char* base = (char*)d_ws;
    size_t off = 0;
    auto carve = [&](size_t bytes) {
        char* p = base + off;
        off += (bytes + 255) & ~(size_t)255;
        return (void*)p;
    };

    unsigned short* qw_bf   = (unsigned short*)carve((size_t)192 * 192 * 2);
    unsigned short* qdw_bf  = (unsigned short*)carve((size_t)192 * 192 * 9 * 2);
    unsigned short* kvw_bf  = (unsigned short*)carve((size_t)384 * 192 * 2);
    unsigned short* projw_bf= (unsigned short*)carve((size_t)192 * 192 * 2);
    unsigned short* pinw_bf = (unsigned short*)carve((size_t)HID * 192 * 2);
    unsigned short* dww_bf  = (unsigned short*)carve((size_t)HID * HID * 9 * 2);
    unsigned short* poutw_bf= (unsigned short*)carve((size_t)192 * HID * 2);
    unsigned short* XN      = (unsigned short*)carve(NCHW * 2);   // LN(x) / later LN(x1)
    unsigned short* ILN     = (unsigned short*)carve(NCHW * 2);   // LN(illum); reused as Q
    unsigned short* Q1      = (unsigned short*)carve(NCHW * 2);   // q 1x1 out; reused as AO
    unsigned short* KV1     = (unsigned short*)carve(NKV * 2);
    unsigned short* KV      = (unsigned short*)carve(NKV * 2);
    float*          ATTNF   = (float*)carve((size_t)8 * 2304 * 4);
    unsigned short* ATTNB   = (unsigned short*)carve((size_t)8 * 2304 * 2);
    unsigned short* H1      = (unsigned short*)carve(NHID * 2);
    unsigned short* H2      = (unsigned short*)carve(NHID * 2);
    unsigned short* Q  = ILN;          // alias: ILN dead after q 1x1
    unsigned short* AO = Q1;           // alias: Q1 dead after q 3x3
    float* X1 = (float*)d_out;         // residual x1 staged in d_out (rewritten by pout)

    // 1) weights -> bf16 (3x3 weights reordered tap-major for the GEMM K order)
    auto cvt = [&](const float* s, unsigned short* d, int n) {
        cvt_f32_bf16_kernel<<<(n + 255) / 256, 256, 0, stream>>>(s, d, n);
    };
    cvt(q_w,    qw_bf,    192 * 192);
    cvt(kv_w,   kvw_bf,   384 * 192);
    cvt(proj_w, projw_bf, 192 * 192);
    cvt(pin_w,  pinw_bf,  HID * 192);
    cvt(pout_w, poutw_bf, 192 * HID);
    cvt3x3_tapmajor_kernel<<<(192 * 192 * 9 + 255) / 256, 256, 0, stream>>>(qdw_w, qdw_bf, 192, 192 * 192 * 9);
    cvt3x3_tapmajor_kernel<<<(HID * HID * 9 + 255) / 256, 256, 0, stream>>>(dw_w, dww_bf, HID, HID * HID * 9);

    // 2) LayerNorms -> bf16
    ln_bf16_kernel<<<(B * HW) / 256, 256, 0, stream>>>(x,     ln1_w, ln1_b, XN,  C);
    ln_bf16_kernel<<<(B * HW) / 256, 256, 0, stream>>>(illum, lnL_w, lnL_b, ILN, C);

    // 3) q = 3x3(1x1(iln))  (full convs via implicit GEMM on WMMA)
    gemm_conv_wmma<<<dim3(128, 3, B), 256, 0, stream>>>(qw_bf,  ILN, nullptr, nullptr, Q1,
                                                        192, 192,  192, 1, 0);
    gemm_conv_wmma<<<dim3(128, 3, B), 256, 0, stream>>>(qdw_bf, Q1,  nullptr, nullptr, Q,
                                                        192, 1728, 192, 9, 0);
    // 4) kv = dw3x3(1x1(xn))
    gemm_conv_wmma<<<dim3(128, 6, B), 256, 0, stream>>>(kvw_bf, XN,  nullptr, nullptr, KV1,
                                                        384, 192,  192, 1, 0);
    dwconv3x3_kernel<<<(int)(NKV / 256), 256, 0, stream>>>(KV1, kvdw_w, KV, 384);

    // 5) L2-normalize q rows and k rows (per batch/head/channel over HW)
    rownorm_kernel<<<384, 256, 0, stream>>>(Q,  192, 192);
    rownorm_kernel<<<384, 256, 0, stream>>>(KV, 192, 384);

    // 6) channel attention: qk^T -> softmax*temp -> attn.v
    qk_wmma_kernel<<<dim3(3, 3, 8), 32, 0, stream>>>(Q, KV, ATTNF);
    softmax_temp_kernel<<<3, 128, 0, stream>>>(ATTNF, temp, ATTNB);
    av_wmma_kernel<<<dim3(1024, 3, 8), 32, 0, stream>>>(ATTNB, KV, AO);

    // 7) x1 = x + proj(out)
    gemm_conv_wmma<<<dim3(128, 3, B), 256, 0, stream>>>(projw_bf, AO, x, X1, nullptr,
                                                        192, 192, 192, 1, 0);

    // 8) FFN: LN -> 1x1 -> full 3x3 + GELU -> 1x1 + residual
    ln_bf16_kernel<<<(B * HW) / 256, 256, 0, stream>>>(X1, ln2_w, ln2_b, XN, C);
    gemm_conv_wmma<<<dim3(128, 8, B), 256, 0, stream>>>(pinw_bf, XN, nullptr, nullptr, H1,
                                                        HID, 192,  192, 1, 0);
    gemm_conv_wmma<<<dim3(128, 8, B), 256, 0, stream>>>(dww_bf,  H1, nullptr, nullptr, H2,
                                                        HID, 4590, HID, 9, 1);
    gemm_conv_wmma<<<dim3(128, 3, B), 256, 0, stream>>>(poutw_bf, H2, X1, (float*)d_out, nullptr,
                                                        192, 510,  HID, 1, 0);
}
